// IntrinsicSignalSynthesizer_59614146069052
// MI455X (gfx1250) — matrix operand order repaired and checked by reading, not compiled, not run on hardware
//
#include <hip/hip_runtime.h>
#include <cfloat>
#include <cmath>

// Problem constants (match reference)
#define BB 16384   // batch
#define DD 512     // feature dim
#define KK 16384   // codebook size
#define HH 256     // hidden
#define ZZ 128     // code dim

typedef __attribute__((ext_vector_type(16))) _Float16 v16h;
typedef __attribute__((ext_vector_type(8)))  _Float16 v8h;
typedef __attribute__((ext_vector_type(8)))  float    v8f;
typedef __attribute__((ext_vector_type(4)))  unsigned int u32x4;
typedef __attribute__((ext_vector_type(8)))  int      i32x8;
typedef __attribute__((ext_vector_type(4)))  int      i32x4;

// ---------------------------------------------------------------------------
// WMMA fragment loaders (f16, 16x16x32)
// A matrix 16x32 (M x K), row-major source with leading dim `ld` (halfs):
//   lane L<16 : row M=L,  K = {k0..k0+15}
//   lane L>=16: row M=L-16, K = {k0+16..k0+31}
__device__ __forceinline__ v16h load_fragA(const _Float16* __restrict__ p, int ld,
                                           int lane, int k0) {
  const int r  = lane & 15;
  const int hi = lane >> 4;
  const _Float16* row = p + (size_t)r * ld + k0 + hi * 16;
  v8h lo = *(const v8h*)(row);
  v8h hv = *(const v8h*)(row + 8);
  v16h out;
#pragma unroll
  for (int i = 0; i < 8; ++i) { out[i] = lo[i]; out[i + 8] = hv[i]; }
  return out;
}

// B matrix 32x16 (K x N), source stored transposed (row-major [N, K], ld halfs):
//   lane L<16 : col N=L,  K = {k0..k0+15} ; lane L>=16: col N=L-16, K = {k0+16..k0+31}
__device__ __forceinline__ v16h load_fragB(const _Float16* __restrict__ p, int ld,
                                           int lane, int k0) {
  const int n  = lane & 15;
  const int hi = lane >> 4;
  const _Float16* col = p + (size_t)n * ld + k0 + hi * 16;
  return *(const v16h*)col;
}

// Same as load_fragB but 16-byte granular (for padded LDS rows, ld in halfs).
__device__ __forceinline__ v16h load_fragB16(const _Float16* p, int ld, int lane, int k0) {
  const int n  = lane & 15;
  const int hi = lane >> 4;
  const _Float16* col = p + (size_t)n * ld + k0 + hi * 16;
  v8h lo = *(const v8h*)(col);
  v8h hv = *(const v8h*)(col + 8);
  v16h out;
#pragma unroll
  for (int i = 0; i < 8; ++i) { out[i] = lo[i]; out[i + 8] = hv[i]; }
  return out;
}

// ---------------------------------------------------------------------------
// Elementwise f32 -> f16 convert
__global__ void conv_f16_kernel(const float* __restrict__ in, _Float16* __restrict__ out, int n) {
  int i = blockIdx.x * 256 + threadIdx.x;
  if (i < n) out[i] = (_Float16)in[i];
}

// Transpose + convert: W [rows, cols] f32 -> Wt [cols, rows] f16
__global__ void transpose_f16_kernel(const float* __restrict__ W, _Float16* __restrict__ Wt,
                                     int rows, int cols) {
  int i = blockIdx.x * 256 + threadIdx.x;
  if (i < rows * cols) {
    int r = i / cols, c = i % cols;
    Wt[(size_t)c * rows + r] = (_Float16)W[i];
  }
}

__global__ void zero_f32_kernel(float* __restrict__ p, int n) {
  int i = blockIdx.x * 256 + threadIdx.x;
  if (i < n) p[i] = 0.0f;
}

// ---------------------------------------------------------------------------
// Generic WMMA GEMM: C[M,N] = A[M,Kd] (f16) @ Bt[N,Kd]^T (f16) + bias
// Block = 128 threads (4 waves); each wave one 16x16 tile. grid = (N/16, M/64)
__global__ void gemm16_kernel(const _Float16* __restrict__ A, const _Float16* __restrict__ Bt,
                              const float* __restrict__ bias, float* __restrict__ C,
                              int N, int Kd) {
  const int lane = threadIdx.x & 31;
  const int wave = threadIdx.x >> 5;
  const int m0 = (blockIdx.y * 4 + wave) * 16;
  const int n0 = blockIdx.x * 16;
  const _Float16* Ab = A + (size_t)m0 * Kd;
  const _Float16* Bb = Bt + (size_t)n0 * Kd;
  v8f acc = {};
  for (int k0 = 0; k0 < Kd; k0 += 32) {
    v16h a = load_fragA(Ab, Kd, lane, k0);
    v16h b = load_fragB(Bb, Kd, lane, k0);
    acc = __builtin_amdgcn_wmma_f32_16x16x32_f16(false, a, false, b, (short)0, acc, false, false);
  }
  const int n  = n0 + (lane & 15);
  const int hi = lane >> 4;
  const float bv = bias[n];
  float* Cr = C + (size_t)(m0 + 8 * hi) * N + n;
#pragma unroll
  for (int vi = 0; vi < 8; ++vi) Cr[(size_t)vi * N] = acc[vi] + bv;
}

// ---------------------------------------------------------------------------
// LayerNorm + exact GELU, row length = HH. One block (256 thr) per row.
__global__ void ln_gelu_kernel(const float* __restrict__ X, const float* __restrict__ g,
                               const float* __restrict__ be, _Float16* __restrict__ Y) {
  __shared__ float red[HH];
  const int row = blockIdx.x;
  const int t = threadIdx.x;
  float x = X[(size_t)row * HH + t];
  red[t] = x;
  __syncthreads();
#pragma unroll
  for (int s = HH / 2; s > 0; s >>= 1) {
    if (t < s) red[t] += red[t + s];
    __syncthreads();
  }
  const float mu = red[0] * (1.0f / HH);
  __syncthreads();
  const float xc = x - mu;
  red[t] = xc * xc;
  __syncthreads();
#pragma unroll
  for (int s = HH / 2; s > 0; s >>= 1) {
    if (t < s) red[t] += red[t + s];
    __syncthreads();
  }
  const float var = red[0] * (1.0f / HH);
  const float y = xc * rsqrtf(var + 1e-5f) * g[t] + be[t];
  const float ge = 0.5f * y * (1.0f + erff(y * 0.70710678118654752f));
  Y[(size_t)row * HH + t] = (_Float16)ge;
}

// ---------------------------------------------------------------------------
// Row squared-norm + f16 convert, row length = ZZ. One block per row.
__global__ void rownorm_kernel(const float* __restrict__ X, float* __restrict__ n2,
                               _Float16* __restrict__ Y) {
  __shared__ float red[ZZ];
  const int row = blockIdx.x;
  const int t = threadIdx.x;
  float x = X[(size_t)row * ZZ + t];
  Y[(size_t)row * ZZ + t] = (_Float16)x;
  red[t] = x * x;
  __syncthreads();
#pragma unroll
  for (int s = ZZ / 2; s > 0; s >>= 1) {
    if (t < s) red[t] += red[t + s];
    __syncthreads();
  }
  if (t == 0) n2[row] = red[0];
}

// ---------------------------------------------------------------------------
// Tensor Data Mover: DMA one codebook tile (NT rows x ZZ halfs, row-major) into
// LDS with 16B padding after every 256B row (pad_interval=5 -> 64 DWORDs,
// pad_amount=3 -> 4 DWORDs) to kill LDS bank conflicts. TENSORcnt tracked.
#define NT 32                 // codebook rows per LDS tile
#define NTILES (KK / NT)      // 512 iterations
#define LDS_ROW 136           // halfs per padded LDS row (272 bytes)

__device__ __forceinline__ void tdm_load_tile(const _Float16* cb, int tile, unsigned lds_addr) {
  const unsigned long long gaddr =
      (unsigned long long)(size_t)(cb + (size_t)tile * NT * ZZ);
  u32x4 g0;
  g0[0] = 1u;                                            // count=1 (valid), user mode
  g0[1] = lds_addr;                                      // lds_addr -> bits [63:32]
  g0[2] = (unsigned)(gaddr & 0xFFFFFFFFu);               // global_addr [95:64]
  g0[3] = (unsigned)((gaddr >> 32) & 0x1FFFFFFu)         // global_addr [120:96]
          | (2u << 30);                                  // type = 2 ("image")
  i32x8 g1;
  g1[0] = (1 << 16)            /* data_size = 2 bytes */
        | (1 << 20)            /* pad_enable           */
        | (5 << 22)            /* pad_interval: 64 DW  */
        | (3 << 25);           /* pad_amount : 4 DW    */
  g1[1] = (ZZ & 0xFFFF) << 16;                           // tensor_dim0[15:0]
  g1[2] = ((ZZ >> 16) & 0xFFFF) | ((KK & 0xFFFF) << 16); // tensor_dim0 hi | tensor_dim1 lo
  g1[3] = ((KK >> 16) & 0xFFFF) | ((ZZ & 0xFFFF) << 16); // tensor_dim1 hi | tile_dim0
  g1[4] = NT;                                            // tile_dim1 = NT, tile_dim2 = 0
  g1[5] = ZZ;                                            // tensor_dim0_stride[31:0]
  g1[6] = 0;                                             // stride0 hi | stride1 lo
  g1[7] = 0;                                             // stride1 hi
  i32x4 gz = {0, 0, 0, 0};
#if defined(__clang_major__) && (__clang_major__ >= 23)
  i32x8 gz8 = {0, 0, 0, 0, 0, 0, 0, 0};
  __builtin_amdgcn_tensor_load_to_lds(g0, g1, gz, gz, gz8, 0);
#else
  __builtin_amdgcn_tensor_load_to_lds(g0, g1, gz, gz, 0);
#endif
}

// ---------------------------------------------------------------------------
// Fused distance + argmin with TDM-fed LDS codebook tiles.
// Block = 256 threads (8 waves), each wave owns a 16-row tile; all 8 waves share
// the double-buffered LDS codebook tile. grid = BB/128.
__global__ void __launch_bounds__(256)
dist_argmin_kernel(const _Float16* __restrict__ E, const float* __restrict__ e2,
                   const _Float16* __restrict__ CB, const float* __restrict__ c2,
                   int* __restrict__ outIdx) {
  __shared__ __align__(16) _Float16 sm[2][NT * LDS_ROW];  // 2 x 8704 B
  const int lane = threadIdx.x & 31;
  const int wave = threadIdx.x >> 5;   // 0..7
  const int m0 = (blockIdx.x * 8 + wave) * 16;
  const int hi = lane >> 4;
  const _Float16* Ab = E + (size_t)m0 * ZZ;

  // Whole A tile (16 x 128) stays in registers for the full K sweep.
  const v16h a0 = load_fragA(Ab, ZZ, lane, 0);
  const v16h a1 = load_fragA(Ab, ZZ, lane, 32);
  const v16h a2 = load_fragA(Ab, ZZ, lane, 64);
  const v16h a3 = load_fragA(Ab, ZZ, lane, 96);

  float e2v[8];
#pragma unroll
  for (int vi = 0; vi < 8; ++vi) e2v[vi] = e2[m0 + 8 * hi + vi];

  float bestd[8];
  int   besti[8];
#pragma unroll
  for (int vi = 0; vi < 8; ++vi) { bestd[vi] = FLT_MAX; besti[vi] = 0; }

  // Prologue: wave 0 kicks off the first DMA.
  if (wave == 0) tdm_load_tile(CB, 0, (unsigned)(size_t)&sm[0][0]);

  for (int it = 0; it < NTILES; ++it) {
    const int b = it & 1;
    __syncthreads();  // previous iteration's reads of sm[b^1] are done
    if (wave == 0) {
      if (it + 1 < NTILES) {
        tdm_load_tile(CB, it + 1, (unsigned)(size_t)&sm[b ^ 1][0]);
        __builtin_amdgcn_s_wait_tensorcnt((short)1);  // tile `it` has landed
      } else {
        __builtin_amdgcn_s_wait_tensorcnt((short)0);
      }
    }
    __syncthreads();  // sm[b] visible to all waves

#pragma unroll
    for (int sub = 0; sub < 2; ++sub) {
      const _Float16* Bb = &sm[b][sub * 16 * LDS_ROW];
      v16h b0 = load_fragB16(Bb, LDS_ROW, lane, 0);
      v16h b1 = load_fragB16(Bb, LDS_ROW, lane, 32);
      v16h b2 = load_fragB16(Bb, LDS_ROW, lane, 64);
      v16h b3 = load_fragB16(Bb, LDS_ROW, lane, 96);
      v8f acc = {};
      acc = __builtin_amdgcn_wmma_f32_16x16x32_f16(false, a0, false, b0, (short)0, acc, false, false);
      acc = __builtin_amdgcn_wmma_f32_16x16x32_f16(false, a1, false, b1, (short)0, acc, false, false);
      acc = __builtin_amdgcn_wmma_f32_16x16x32_f16(false, a2, false, b2, (short)0, acc, false, false);
      acc = __builtin_amdgcn_wmma_f32_16x16x32_f16(false, a3, false, b3, (short)0, acc, false, false);

      const int nidx = it * NT + sub * 16 + (lane & 15);
      const float c2n = c2[nidx];
#pragma unroll
      for (int vi = 0; vi < 8; ++vi) {
        float d2 = e2v[vi] + c2n - 2.0f * acc[vi];
        if (d2 < bestd[vi]) { bestd[vi] = d2; besti[vi] = nidx; }
      }
    }
  }

  // Reduce across the 16 lanes (columns) of each half-wave, per row.
#pragma unroll
  for (int vi = 0; vi < 8; ++vi) {
    float bv = bestd[vi];
    int   bi = besti[vi];
#pragma unroll
    for (int mask = 1; mask < 16; mask <<= 1) {
      float ov = __shfl_xor(bv, mask, 32);
      int   oi = __shfl_xor(bi, mask, 32);
      if (ov < bv || (ov == bv && oi < bi)) { bv = ov; bi = oi; }
    }
    if ((lane & 15) == 0) outIdx[m0 + 8 * hi + vi] = bi;
  }
}

// ---------------------------------------------------------------------------
// Gather quantized codes (f16) by index. One block (128 thr) per batch row.
__global__ void gather_kernel(const _Float16* __restrict__ CB, const int* __restrict__ idx,
                              _Float16* __restrict__ Q) {
  const int b = blockIdx.x;
  const int t = threadIdx.x;
  Q[(size_t)b * ZZ + t] = CB[(size_t)idx[b] * ZZ + t];
}

// ---------------------------------------------------------------------------
// Final GEMM fused with squared reconstruction error accumulation.
// A [BB, HH] f16, Bt [DD, HH] f16, feat [BB, DD] f32. grid = (DD/16, BB/64)
__global__ void gemm_err_kernel(const _Float16* __restrict__ A, const _Float16* __restrict__ Bt,
                                const float* __restrict__ bias, const float* __restrict__ feat,
                                float* __restrict__ errsum) {
  const int lane = threadIdx.x & 31;
  const int wave = threadIdx.x >> 5;
  const int m0 = (blockIdx.y * 4 + wave) * 16;
  const int n0 = blockIdx.x * 16;
  const _Float16* Ab = A + (size_t)m0 * HH;
  const _Float16* Bb = Bt + (size_t)n0 * HH;
  v8f acc = {};
#pragma unroll
  for (int k0 = 0; k0 < HH; k0 += 32) {
    v16h a = load_fragA(Ab, HH, lane, k0);
    v16h b = load_fragB(Bb, HH, lane, k0);
    acc = __builtin_amdgcn_wmma_f32_16x16x32_f16(false, a, false, b, (short)0, acc, false, false);
  }
  const int n  = n0 + (lane & 15);
  const int hi = lane >> 4;
  const float bv = bias[n];
#pragma unroll
  for (int vi = 0; vi < 8; ++vi) {
    const int m = m0 + 8 * hi + vi;
    float d = acc[vi] + bv - feat[(size_t)m * DD + n];
    float sq = d * d;
#pragma unroll
    for (int mask = 1; mask < 16; mask <<= 1) sq += __shfl_xor(sq, mask, 32);
    if ((lane & 15) == 0) atomicAdd(&errsum[m], sq);
  }
}

// ---------------------------------------------------------------------------
__global__ void err_mean_kernel(const float* __restrict__ errsum, float* __restrict__ outErr,
                                float* __restrict__ meanacc) {
  __shared__ float red[256];
  const int i = blockIdx.x * 256 + threadIdx.x;
  float e = errsum[i] * (1.0f / DD);
  outErr[i] = e;
  red[threadIdx.x] = e;
  __syncthreads();
#pragma unroll
  for (int s = 128; s > 0; s >>= 1) {
    if (threadIdx.x < s) red[threadIdx.x] += red[threadIdx.x + s];
    __syncthreads();
  }
  if (threadIdx.x == 0) atomicAdd(meanacc, red[0]);
}

__global__ void finalize_kernel(const float* __restrict__ err, const float* __restrict__ meanacc,
                                const int* __restrict__ idx, float* __restrict__ out) {
  const int i = blockIdx.x * 256 + threadIdx.x;
  const float scale = meanacc[0] * (1.0f / BB) + 1e-8f;
  const float lp = -fabsf(err[i]) / scale - logf(2.0f * scale);
  const float ebits = -lp * 1.4426950408889634f;  // / ln 2
  const float tb = 14.0f + ebits;                 // log2(K) = 14
  out[BB + i]     = (DD * 32.0f) / tb;
  out[2 * BB + i] = tb;
  out[3 * BB + i] = (float)idx[i];
}

// ---------------------------------------------------------------------------
extern "C" void kernel_launch(void* const* d_in, const int* in_sizes, int n_in,
                              void* d_out, int out_size, void* d_ws, size_t ws_size,
                              hipStream_t stream) {
  const float* features = (const float*)d_in[0];
  const float* W1 = (const float*)d_in[1];
  const float* b1 = (const float*)d_in[2];
  const float* g1 = (const float*)d_in[3];
  const float* be1 = (const float*)d_in[4];
  const float* W2 = (const float*)d_in[5];
  const float* b2 = (const float*)d_in[6];
  const float* codebook = (const float*)d_in[7];
  const float* W3 = (const float*)d_in[8];
  const float* b3 = (const float*)d_in[9];
  const float* g2 = (const float*)d_in[10];
  const float* be2 = (const float*)d_in[11];
  const float* W4 = (const float*)d_in[12];
  const float* b4 = (const float*)d_in[13];
  float* out = (float*)d_out;

  char* ws = (char*)d_ws;
  size_t off = 0;
  auto take = [&](size_t bytes) { char* p = ws + off; off += (bytes + 255) & ~(size_t)255; return p; };
  _Float16* featf16 = (_Float16*)take((size_t)BB * DD * 2);
  _Float16* cbf16   = (_Float16*)take((size_t)KK * ZZ * 2);
  _Float16* W1t     = (_Float16*)take((size_t)HH * DD * 2);
  _Float16* W2t     = (_Float16*)take((size_t)ZZ * HH * 2);
  _Float16* W3t     = (_Float16*)take((size_t)HH * ZZ * 2);
  _Float16* W4t     = (_Float16*)take((size_t)DD * HH * 2);
  float*    tmp32   = (float*)take((size_t)BB * HH * 4);
  _Float16* hf16    = (_Float16*)take((size_t)BB * HH * 2);
  float*    Ef32    = (float*)take((size_t)BB * ZZ * 4);
  _Float16* ef16    = (_Float16*)take((size_t)BB * ZZ * 2);
  float*    e2      = (float*)take((size_t)BB * 4);
  float*    c2      = (float*)take((size_t)KK * 4);
  int*      idxbuf  = (int*)take((size_t)BB * 4);
  _Float16* qf16    = (_Float16*)take((size_t)BB * ZZ * 2);
  float*    errsum  = (float*)take((size_t)(BB + 256) * 4);
  float*    meanacc = errsum + BB;

  // 1) Precision conversion / weight pre-transposition
  conv_f16_kernel<<<(BB * DD + 255) / 256, 256, 0, stream>>>(features, featf16, BB * DD);
  transpose_f16_kernel<<<(DD * HH + 255) / 256, 256, 0, stream>>>(W1, W1t, DD, HH);
  transpose_f16_kernel<<<(HH * ZZ + 255) / 256, 256, 0, stream>>>(W2, W2t, HH, ZZ);
  transpose_f16_kernel<<<(ZZ * HH + 255) / 256, 256, 0, stream>>>(W3, W3t, ZZ, HH);
  transpose_f16_kernel<<<(HH * DD + 255) / 256, 256, 0, stream>>>(W4, W4t, HH, DD);
  rownorm_kernel<<<KK, ZZ, 0, stream>>>(codebook, c2, cbf16);

  // 2) Encoder
  gemm16_kernel<<<dim3(HH / 16, BB / 64), 128, 0, stream>>>(featf16, W1t, b1, tmp32, HH, DD);
  ln_gelu_kernel<<<BB, HH, 0, stream>>>(tmp32, g1, be1, hf16);
  gemm16_kernel<<<dim3(ZZ / 16, BB / 64), 128, 0, stream>>>(hf16, W2t, b2, Ef32, ZZ, HH);
  rownorm_kernel<<<BB, ZZ, 0, stream>>>(Ef32, e2, ef16);

  // 3) Vector quantization: TDM-fed distances + argmin (dominant compute)
  dist_argmin_kernel<<<BB / 128, 256, 0, stream>>>(ef16, e2, cbf16, c2, idxbuf);
  gather_kernel<<<BB, ZZ, 0, stream>>>(cbf16, idxbuf, qf16);

  // 4) Decoder + fused reconstruction error
  gemm16_kernel<<<dim3(HH / 16, BB / 64), 128, 0, stream>>>(qf16, W3t, b3, tmp32, HH, ZZ);
  ln_gelu_kernel<<<BB, HH, 0, stream>>>(tmp32, g2, be2, hf16);
  zero_f32_kernel<<<(BB + 256 + 255) / 256, 256, 0, stream>>>(errsum, BB + 1);
  gemm_err_kernel<<<dim3(DD / 16, BB / 64), 128, 0, stream>>>(hf16, W4t, b4, features, errsum);

  // 5) MDL outputs
  err_mean_kernel<<<BB / 256, 256, 0, stream>>>(errsum, out, meanacc);
  finalize_kernel<<<BB / 256, 256, 0, stream>>>(out, meanacc, idxbuf, out);
}
